// MPNNPositionProducer_42279658061911
// MI455X (gfx1250) — compile-verified, matrix operand order checked by reading, not run on hardware
//
#include <hip/hip_runtime.h>
#include <hip/hip_bf16.h>

typedef _Float16 h16;
typedef __attribute__((ext_vector_type(16))) _Float16 v16h;
typedef __attribute__((ext_vector_type(8)))  _Float16 v8h;
typedef __attribute__((ext_vector_type(8)))  float    v8f;

#define NN   4096
#define MM   8192
#define NDIMK  64
#define EDIMK  16
#define HH     64
#define CCH    64
#define LLAY    4

__device__ __forceinline__ float lrelu_f(float x) { return x > 0.f ? x : 0.01f * x; }

union AFrag { v16h v; v8h h[2]; };

// Load one 16x32 f16 WMMA fragment row-chunk for this lane.
// rowptr points at this lane's matrix row (row m = lane&15 for A, col n = lane&15 for B^T),
// kh = lane>>4. ISA layout: vgprs0-3 hold K[8*kh..8*kh+7], vgprs4-7 hold K[16+8*kh..+7].
__device__ __forceinline__ v16h load_frag(const h16* __restrict__ rowptr, int kh) {
  AFrag u;
  u.h[0] = *(const v8h*)(rowptr + 8 * kh);
  u.h[1] = *(const v8h*)(rowptr + 16 + 8 * kh);
  return u.v;
}

// ---------------------------------------------------------------------------
// Generic f16 WMMA GEMM:  out = act(A[Mr x K] @ Bt^T + bias)
// Bt stored transposed: [Nout, K] f16 row-major. One wave -> 16 rows x 64 cols.
// grid.x = Mr/128 (8 waves/block), grid.y = Nout/64.
// ---------------------------------------------------------------------------
__global__ __launch_bounds__(256)
void gemm_f16_wmma(const h16* __restrict__ A, int lda,
                   const h16* __restrict__ Bt,
                   const float* __restrict__ bias,
                   float* __restrict__ outF32, h16* __restrict__ outF16,
                   int Nout, int K, int do_lrelu)
{
  const int wave = threadIdx.x >> 5;
  const int lane = threadIdx.x & 31;
  const int l15 = lane & 15;
  const int kh = lane >> 4;
  const int row0 = (blockIdx.x * 8 + wave) * 16;
  const int ncb = blockIdx.y * 64;

  v8f acc0 = {}, acc1 = {}, acc2 = {}, acc3 = {};
  const h16* arow = A + (size_t)(row0 + l15) * lda;
  const h16* b0r = Bt + (size_t)(ncb + 0 * 16 + l15) * K;
  const h16* b1r = Bt + (size_t)(ncb + 1 * 16 + l15) * K;
  const h16* b2r = Bt + (size_t)(ncb + 2 * 16 + l15) * K;
  const h16* b3r = Bt + (size_t)(ncb + 3 * 16 + l15) * K;

  for (int kt = 0; kt < K; kt += 32) {
    v16h a = load_frag(arow + kt, kh);
    v16h b0 = load_frag(b0r + kt, kh);
    v16h b1 = load_frag(b1r + kt, kh);
    v16h b2 = load_frag(b2r + kt, kh);
    v16h b3 = load_frag(b3r + kt, kh);
    acc0 = __builtin_amdgcn_wmma_f32_16x16x32_f16(false, a, false, b0, (short)0, acc0, false, false);
    acc1 = __builtin_amdgcn_wmma_f32_16x16x32_f16(false, a, false, b1, (short)0, acc1, false, false);
    acc2 = __builtin_amdgcn_wmma_f32_16x16x32_f16(false, a, false, b2, (short)0, acc2, false, false);
    acc3 = __builtin_amdgcn_wmma_f32_16x16x32_f16(false, a, false, b3, (short)0, acc3, false, false);
  }

  v8f accs[4] = { acc0, acc1, acc2, acc3 };
#pragma unroll
  for (int nt = 0; nt < 4; nt++) {
    int c = ncb + nt * 16 + l15;
    float bb = bias ? bias[c] : 0.f;
#pragma unroll
    for (int v = 0; v < 8; v++) {
      int r = row0 + 8 * kh + v;   // C/D layout: vgpr v holds row 8*kh + v
      float x = accs[nt][v] + bb;
      if (do_lrelu) x = lrelu_f(x);
      if (outF32) outF32[(size_t)r * Nout + c] = x;
      if (outF16) outF16[(size_t)r * Nout + c] = (h16)x;
    }
  }
}

// ---------------------------------------------------------------------------
// Message GEMM: hid = lrelu(concat(h[us], h[vs], e) @ W1 + b1)
// A rows are gathered per edge across K = 192 (3 x 64 segments).
// ---------------------------------------------------------------------------
__global__ __launch_bounds__(256)
void hid_wmma_kernel(const h16* __restrict__ hbuf, const h16* __restrict__ ebuf,
                     const int* __restrict__ us, const int* __restrict__ vs,
                     const h16* __restrict__ W1t, const float* __restrict__ b1,
                     float* __restrict__ hidF32, h16* __restrict__ hid16)
{
  const int wave = threadIdx.x >> 5;
  const int lane = threadIdx.x & 31;
  const int l15 = lane & 15;
  const int kh = lane >> 4;
  const int row0 = (blockIdx.x * 8 + wave) * 16;
  const int edge = row0 + l15;
  const int u = us[edge];
  const int v = vs[edge];
  const h16* hu = hbuf + (size_t)u * 64;
  const h16* hv = hbuf + (size_t)v * 64;
  const h16* ee = ebuf + (size_t)edge * 64;

  v8f acc0 = {}, acc1 = {}, acc2 = {}, acc3 = {};
#pragma unroll
  for (int kt = 0; kt < 6; kt++) {
    const h16* arow;
    if (kt < 2)      arow = hu + kt * 32;
    else if (kt < 4) arow = hv + (kt - 2) * 32;
    else             arow = ee + (kt - 4) * 32;
    v16h a = load_frag(arow, kh);
    v16h b0 = load_frag(W1t + (size_t)(0 * 16 + l15) * 192 + kt * 32, kh);
    v16h b1v = load_frag(W1t + (size_t)(1 * 16 + l15) * 192 + kt * 32, kh);
    v16h b2v = load_frag(W1t + (size_t)(2 * 16 + l15) * 192 + kt * 32, kh);
    v16h b3v = load_frag(W1t + (size_t)(3 * 16 + l15) * 192 + kt * 32, kh);
    acc0 = __builtin_amdgcn_wmma_f32_16x16x32_f16(false, a, false, b0, (short)0, acc0, false, false);
    acc1 = __builtin_amdgcn_wmma_f32_16x16x32_f16(false, a, false, b1v, (short)0, acc1, false, false);
    acc2 = __builtin_amdgcn_wmma_f32_16x16x32_f16(false, a, false, b2v, (short)0, acc2, false, false);
    acc3 = __builtin_amdgcn_wmma_f32_16x16x32_f16(false, a, false, b3v, (short)0, acc3, false, false);
  }

  v8f accs[4] = { acc0, acc1, acc2, acc3 };
#pragma unroll
  for (int nt = 0; nt < 4; nt++) {
    int c = nt * 16 + l15;
    float bb = b1[c];
#pragma unroll
    for (int vv = 0; vv < 8; vv++) {
      int r = row0 + 8 * kh + vv;
      float x = lrelu_f(accs[nt][vv] + bb);
      hidF32[(size_t)r * 64 + c] = x;
      hid16[(size_t)r * 64 + c] = (h16)x;
    }
  }
}

// ------------------------- small helper kernels ----------------------------
__global__ void f2h_kernel(const float* __restrict__ x, h16* __restrict__ y, int n) {
  int i = blockIdx.x * blockDim.x + threadIdx.x;
  if (i < n) y[i] = (h16)x[i];
}

// W [K, Nout] f32 row-major -> Wt [Nout, K] f16 row-major
__global__ void wconv_kernel(const float* __restrict__ W, h16* __restrict__ Wt, int K, int Nout) {
  int idx = blockIdx.x * blockDim.x + threadIdx.x;
  if (idx >= K * Nout) return;
  int n = idx / K, k = idx % K;
  Wt[idx] = (h16)W[(size_t)k * Nout + n];
}

// e = lrelu(edge_features @ FCE_W + FCE_b), K=16 -> plain VALU (tiny)
__global__ void eproj_kernel(const float* __restrict__ ef, const float* __restrict__ W,
                             const float* __restrict__ b, h16* __restrict__ e16) {
  int idx = blockIdx.x * blockDim.x + threadIdx.x; // MM*64
  int j = idx >> 6, c = idx & 63;
  float s = b[c];
#pragma unroll
  for (int k = 0; k < 16; k++) s += ef[(size_t)j * 16 + k] * W[(size_t)k * 64 + c];
  e16[idx] = (h16)lrelu_f(s);
}

__global__ void att_kernel(const float* __restrict__ hidF32, const float* __restrict__ wa,
                           const float* __restrict__ ba, float* __restrict__ att) {
  int j = blockIdx.x * blockDim.x + threadIdx.x;
  if (j >= MM) return;
  float s = ba[0];
  const float* row = hidF32 + (size_t)j * 64;
#pragma unroll 8
  for (int c = 0; c < 64; c++) s += row[c] * wa[c];
  att[j] = lrelu_f(s);
}

__global__ void colmean_kernel(const float* __restrict__ hidF32, float* __restrict__ cm) {
  int c = threadIdx.x; // 64 threads, 1 block
  float s = 0.f;
  for (int j = 0; j < MM; j++) s += hidF32[(size_t)j * 64 + c];
  cm[c] = s * (1.0f / MM);
}

// CSR build (deterministic: rank = #earlier edges into same node)
__global__ void count_kernel(const int* __restrict__ vs, int* __restrict__ deg) {
  int j = blockIdx.x * blockDim.x + threadIdx.x;
  if (j < MM) atomicAdd(&deg[vs[j]], 1);
}

__global__ void scan_kernel(const int* __restrict__ deg, int* __restrict__ off) {
  __shared__ int sh[1024];
  int t = threadIdx.x;
  int v0 = deg[4 * t], v1 = deg[4 * t + 1], v2 = deg[4 * t + 2], v3 = deg[4 * t + 3];
  int s = v0 + v1 + v2 + v3;
  sh[t] = s;
  __syncthreads();
  for (int d = 1; d < 1024; d <<= 1) {
    int x = (t >= d) ? sh[t - d] : 0;
    __syncthreads();
    sh[t] += x;
    __syncthreads();
  }
  int excl = sh[t] - s;
  off[4 * t] = excl;
  off[4 * t + 1] = excl + v0;
  off[4 * t + 2] = excl + v0 + v1;
  off[4 * t + 3] = excl + v0 + v1 + v2;
}

__global__ void fill_kernel(const int* __restrict__ vs, const int* __restrict__ off,
                            int* __restrict__ elist) {
  int j = blockIdx.x * blockDim.x + threadIdx.x;
  if (j >= MM) return;
  int v = vs[j], r = 0;
  for (int k = 0; k < j; k++) r += (vs[k] == v) ? 1 : 0;
  elist[off[v] + r] = j;
}

// Segment softmax + weighted sum. Block = 64 threads (one per channel), grid = N.
// deg==0 -> reference softmaxes an all-NEG row -> uniform 1/M -> column mean of hid.
__global__ void context_kernel(const int* __restrict__ deg, const int* __restrict__ off,
                               const int* __restrict__ elist, const float* __restrict__ att,
                               const float* __restrict__ hidF32, const float* __restrict__ cm,
                               float* __restrict__ ctxF32, h16* __restrict__ ctx16) {
  int n = blockIdx.x, c = threadIdx.x;
  int d = deg[n];
  float out;
  if (d == 0) {
    out = cm[c];
  } else {
    int base = off[n];
    float m = -3.402823466e38f;
    for (int t = 0; t < d; t++) m = fmaxf(m, att[elist[base + t]]);
    float s = 0.f, acc = 0.f;
    for (int t = 0; t < d; t++) {
      int e = elist[base + t];
      float w = expf(att[e] - m);
      s += w;
      acc += w * hidF32[(size_t)e * 64 + c];
    }
    out = acc / s;
  }
  ctxF32[(size_t)n * 64 + c] = out;
  ctx16[(size_t)n * 64 + c] = (h16)out;
}

__global__ void gru_kernel(const float* __restrict__ gi, const float* __restrict__ gh,
                           const float* __restrict__ hin, float* __restrict__ hout,
                           h16* __restrict__ h16out, int do_relu) {
  int idx = blockIdx.x * blockDim.x + threadIdx.x; // NN*64
  int n = idx >> 6, c = idx & 63;
  const float* gir = gi + (size_t)n * 192;
  const float* ghr = gh + (size_t)n * 192;
  float r = 1.f / (1.f + expf(-(gir[c] + ghr[c])));
  float z = 1.f / (1.f + expf(-(gir[64 + c] + ghr[64 + c])));
  float nn = tanhf(gir[128 + c] + r * ghr[128 + c]);
  float hv = (1.f - z) * nn + z * hin[idx];
  if (do_relu) hv = fmaxf(hv, 0.f);
  hout[idx] = hv;
  if (h16out) h16out[idx] = (h16)hv;
}

// ---------------------------------------------------------------------------
extern "C" void kernel_launch(void* const* d_in, const int* in_sizes, int n_in,
                              void* d_out, int out_size, void* d_ws, size_t ws_size,
                              hipStream_t stream) {
  (void)in_sizes; (void)n_in; (void)out_size; (void)ws_size;
  const float* node_features = (const float*)d_in[0];
  const float* edge_features = (const float*)d_in[1];
  const int*   us  = (const int*)d_in[2];
  const int*   vs  = (const int*)d_in[3];
  // d_in[4] node_edge_matrix, d_in[5] node_edge_mask: intentionally unused
  const float* FCN_W = (const float*)d_in[6];
  const float* FCN_b = (const float*)d_in[7];
  const float* FCE_W = (const float*)d_in[8];
  const float* FCE_b = (const float*)d_in[9];
  const float* W1  = (const float*)d_in[10];
  const float* b1  = (const float*)d_in[11];
  const float* Wa  = (const float*)d_in[12];
  const float* ba  = (const float*)d_in[13];
  const float* W2  = (const float*)d_in[14];
  const float* b2  = (const float*)d_in[15];
  const float* Wih = (const float*)d_in[16];
  const float* Whh = (const float*)d_in[17];
  const float* bih = (const float*)d_in[18];
  const float* bhh = (const float*)d_in[19];

  char* ws = (char*)d_ws;
  size_t o = 0;
  auto alloc = [&](size_t bytes) -> char* {
    char* p = ws + o;
    o = (o + bytes + 255) & ~(size_t)255;
    return p;
  };
  h16*   nf16  = (h16*)alloc((size_t)NN * 64 * 2);
  h16*   FCNt  = (h16*)alloc(64 * 64 * 2);
  h16*   W1t   = (h16*)alloc((size_t)LLAY * 64 * 192 * 2);
  h16*   W2t   = (h16*)alloc((size_t)LLAY * 64 * 64 * 2);
  h16*   Wiht  = (h16*)alloc((size_t)LLAY * 192 * 64 * 2);
  h16*   Whht  = (h16*)alloc((size_t)LLAY * 192 * 64 * 2);
  float* hF    = (float*)alloc((size_t)NN * 64 * 4);
  h16*   h16b  = (h16*)alloc((size_t)NN * 64 * 2);
  h16*   e16b  = (h16*)alloc((size_t)MM * 64 * 2);
  float* hidF  = (float*)alloc((size_t)MM * 64 * 4);
  h16*   hid16 = (h16*)alloc((size_t)MM * 64 * 2);
  float* attb  = (float*)alloc((size_t)MM * 4);
  int*   deg   = (int*)alloc((size_t)NN * 4);
  int*   off   = (int*)alloc((size_t)NN * 4);
  int*   elist = (int*)alloc((size_t)MM * 4);
  float* cmean = (float*)alloc(64 * 4);
  float* ctxF  = (float*)alloc((size_t)NN * 64 * 4);
  h16*   ctx16 = (h16*)alloc((size_t)NN * 64 * 2);
  float* gi    = (float*)alloc((size_t)NN * 192 * 4);
  float* gh    = (float*)alloc((size_t)NN * 192 * 4);

  // ---- weight / activation conversion (f32 -> f16, weights transposed) ----
  f2h_kernel<<<(NN * 64 + 255) / 256, 256, 0, stream>>>(node_features, nf16, NN * 64);
  wconv_kernel<<<(64 * 64 + 255) / 256, 256, 0, stream>>>(FCN_W, FCNt, 64, 64);
  for (int i = 0; i < LLAY; i++) {
    wconv_kernel<<<(192 * 64 + 255) / 256, 256, 0, stream>>>(W1 + (size_t)i * 192 * 64, W1t + (size_t)i * 64 * 192, 192, 64);
    wconv_kernel<<<(64 * 64 + 255) / 256, 256, 0, stream>>>(W2 + (size_t)i * 64 * 64, W2t + (size_t)i * 64 * 64, 64, 64);
    wconv_kernel<<<(64 * 192 + 255) / 256, 256, 0, stream>>>(Wih + (size_t)i * 64 * 192, Wiht + (size_t)i * 192 * 64, 64, 192);
    wconv_kernel<<<(64 * 192 + 255) / 256, 256, 0, stream>>>(Whh + (size_t)i * 64 * 192, Whht + (size_t)i * 192 * 64, 64, 192);
  }

  // ---- CSR build from vs (deterministic) ----
  hipMemsetAsync(deg, 0, (size_t)NN * 4, stream);
  count_kernel<<<MM / 256, 256, 0, stream>>>(vs, deg);
  scan_kernel<<<1, 1024, 0, stream>>>(deg, off);
  fill_kernel<<<MM / 256, 256, 0, stream>>>(vs, off, elist);

  // ---- input projections ----
  gemm_f16_wmma<<<dim3(NN / 128, 1), 256, 0, stream>>>(nf16, 64, FCNt, FCN_b, hF, h16b, 64, 64, 1);
  eproj_kernel<<<(MM * 64) / 256, 256, 0, stream>>>(edge_features, FCE_W, FCE_b, e16b);

  // ---- message passing layers ----
  for (int i = 0; i < LLAY; i++) {
    hid_wmma_kernel<<<MM / 128, 256, 0, stream>>>(h16b, e16b, us, vs,
                                                  W1t + (size_t)i * 64 * 192, b1 + (size_t)i * 64,
                                                  hidF, hid16);
    att_kernel<<<MM / 256, 256, 0, stream>>>(hidF, Wa + (size_t)i * 64, ba + i, attb);
    colmean_kernel<<<1, 64, 0, stream>>>(hidF, cmean);
    context_kernel<<<NN, 64, 0, stream>>>(deg, off, elist, attb, hidF, cmean, ctxF, ctx16);
    // new_e (f16 only; overwrites e16b after hid consumed it)
    gemm_f16_wmma<<<dim3(MM / 128, 1), 256, 0, stream>>>(hid16, 64, W2t + (size_t)i * 64 * 64,
                                                         b2 + (size_t)i * 64, nullptr, e16b, 64, 64, 1);
    // GRU gate GEMMs
    gemm_f16_wmma<<<dim3(NN / 128, 3), 256, 0, stream>>>(ctx16, 64, Wiht + (size_t)i * 192 * 64,
                                                         bih + (size_t)i * 192, gi, nullptr, 192, 64, 0);
    gemm_f16_wmma<<<dim3(NN / 128, 3), 256, 0, stream>>>(h16b, 64, Whht + (size_t)i * 192 * 64,
                                                         bhh + (size_t)i * 192, gh, nullptr, 192, 64, 0);
    int last = (i == LLAY - 1);
    gru_kernel<<<(NN * 64) / 256, 256, 0, stream>>>(gi, gh, hF,
                                                    last ? (float*)d_out : hF,
                                                    last ? nullptr : h16b,
                                                    last ? 0 : 1);
  }
}
